// PatientPoolEncoder_50328426775179
// MI455X (gfx1250) — compile-verified, matrix operand order checked by reading, not run on hardware
//
#include <hip/hip_runtime.h>
#include <hip/hip_bf16.h>

// ---------------- problem constants ----------------
#define NB    32
#define LSEQ  2048
#define CCH   128
#define NIN   3
#define OUTC  1172             // 3*(3*128) + 20
#define CH    32               // L-chunk length for the chunked scan
#define NCH   (LSEQ / CH)      // 64 chunks
#define CGRP  8                // chunks per block (1 wave per chunk)
#define NROWS (NIN * NB * CCH) // 12288 independent scan rows

typedef __attribute__((ext_vector_type(16))) _Float16 v16h;
typedef __attribute__((ext_vector_type(8)))  float    v8f;
typedef __attribute__((ext_vector_type(4)))  float    f4;
typedef __attribute__((ext_vector_type(4)))  int      i4;

// =====================================================================
// MLP: d = relu(relu(dem@W1+b1)@W2+b2), [32,8]->[32,40]->[32,20]
// One wave32; v_wmma_f32_16x16x32_f16 with ISA-documented layouts.
// Fully unrolled so all register arrays index statically (no v_movrels).
// =====================================================================
__global__ __launch_bounds__(32) void mlp_wmma(const float* __restrict__ dem,
                                               const float* __restrict__ W1,
                                               const float* __restrict__ b1,
                                               const float* __restrict__ W2,
                                               const float* __restrict__ b2,
                                               float* __restrict__ d20) {
    const int lane  = threadIdx.x;
    const int mlo   = lane & 15;
    const int kbase = (lane < 16) ? 0 : 8;
    const int mofs  = (lane < 16) ? 0 : 8;
    __shared__ float hsh[32][48];

    // ---- layer 1: A = dem (K=8 zero-padded to 32), B = W1 (N=40 padded to 48)
    v16h A[2];
    v16h B[3];
#pragma unroll
    for (int p = 0; p < 16; ++p) {
        const int kl = ((p < 8) ? p : p + 8) + kbase;   // K index for this half
#pragma unroll
        for (int i = 0; i < 2; ++i)
            A[i][p] = (kl < 8) ? (_Float16)dem[(i * 16 + mlo) * 8 + kl] : (_Float16)0.f;
#pragma unroll
        for (int j = 0; j < 3; ++j) {
            const int n = j * 16 + mlo;
            B[j][p] = (kl < 8 && n < 40) ? (_Float16)W1[kl * 40 + n] : (_Float16)0.f;
        }
    }
#pragma unroll
    for (int i = 0; i < 2; ++i)
#pragma unroll
        for (int j = 0; j < 3; ++j) {
            v8f acc;
#pragma unroll
            for (int r = 0; r < 8; ++r) acc[r] = 0.f;
            acc = __builtin_amdgcn_wmma_f32_16x16x32_f16(false, A[i], false, B[j],
                                                         (short)0, acc, false, false);
            const int col = j * 16 + mlo;
#pragma unroll
            for (int r = 0; r < 8; ++r) {
                const int m = i * 16 + mofs + r;
                const float v = acc[r] + ((col < 40) ? b1[col] : 0.f);
                hsh[m][col] = fmaxf(v, 0.f);
            }
        }
    __syncthreads();

    // ---- layer 2: A2 = h [32,40] (K padded to 64), B2 = W2 (N=20 padded to 32)
    v16h A2[2][2];
    v16h B2[2][2];
#pragma unroll
    for (int p = 0; p < 16; ++p) {
        const int kl = ((p < 8) ? p : p + 8) + kbase;
#pragma unroll
        for (int kc = 0; kc < 2; ++kc) {
            const int k = kc * 32 + kl;
#pragma unroll
            for (int i = 0; i < 2; ++i)
                A2[i][kc][p] = (k < 40) ? (_Float16)hsh[i * 16 + mlo][k] : (_Float16)0.f;
#pragma unroll
            for (int jn = 0; jn < 2; ++jn) {
                const int n = jn * 16 + mlo;
                B2[jn][kc][p] = (k < 40 && n < 20) ? (_Float16)W2[k * 20 + n] : (_Float16)0.f;
            }
        }
    }
#pragma unroll
    for (int i = 0; i < 2; ++i)
#pragma unroll
        for (int jn = 0; jn < 2; ++jn) {
            v8f acc;
#pragma unroll
            for (int r = 0; r < 8; ++r) acc[r] = 0.f;
            acc = __builtin_amdgcn_wmma_f32_16x16x32_f16(false, A2[i][0], false, B2[jn][0],
                                                         (short)0, acc, false, false);
            acc = __builtin_amdgcn_wmma_f32_16x16x32_f16(false, A2[i][1], false, B2[jn][1],
                                                         (short)0, acc, false, false);
            const int col = jn * 16 + mlo;
            if (col < 20)
#pragma unroll
                for (int r = 0; r < 8; ++r) {
                    const int m = i * 16 + mofs + r;
                    d20[m * 20 + col] = fmaxf(acc[r] + b2[col], 0.f);
                }
        }
}

__device__ __forceinline__ const float* selInput(int in, const float* i0,
                                                 const float* i1, const float* i2) {
    return (in == 0) ? i0 : (in == 1) ? i1 : i2;
}

// =====================================================================
// Phase A: per-(row, chunk) aggregates (sum, max, argmax).
// 1 wave per (input, n, chunk); lane = channel quad -> B128 loads,
// 512B contiguous per wave per step. Aggregates laid out [chunk][row]
// so the quad store is one B128 and phase B's walk is coalesced.
// grid = NIN*NB*(NCH/CGRP) blocks x 256 threads (8 waves = 8 chunks)
// =====================================================================
__global__ __launch_bounds__(256) void phaseA(const float* __restrict__ i0,
                                              const float* __restrict__ i1,
                                              const float* __restrict__ i2,
                                              f4* __restrict__ aggSum,
                                              f4* __restrict__ aggMax,
                                              i4* __restrict__ aggIdx) {
    const int b   = blockIdx.x;
    const int cg  = b % (NCH / CGRP);
    const int n   = (b / (NCH / CGRP)) % NB;
    const int in  = b / ((NCH / CGRP) * NB);
    const int c4  = threadIdx.x & 31;                 // channel quad 0..31
    const int chunk = cg * CGRP + (threadIdx.x >> 5); // wave id -> chunk
    const int t0 = chunk * CH;

    const float* x = selInput(in, i0, i1, i2) + (size_t)n * LSEQ * CCH + c4 * 4;

    f4 s  = {0.f, 0.f, 0.f, 0.f};
    f4 m  = {-INFINITY, -INFINITY, -INFINITY, -INFINITY};
    i4 am = {0, 0, 0, 0};
    for (int tt = 0; tt < CH; ++tt) {
        __builtin_prefetch(&x[(size_t)(t0 + tt + 8) * CCH], 0, 1);  // global_prefetch_b8
        const f4 v = *(const f4*)(x + (size_t)(t0 + tt) * CCH);
#pragma unroll
        for (int e = 0; e < 4; ++e) {
            if (v[e] > m[e]) { m[e] = v[e]; am[e] = t0 + tt; }  // strict >: earliest wins
            s[e] += v[e];
        }
    }
    const int row = ((in * NB + n) * CCH + c4 * 4);   // multiple of 4
    const size_t q = ((size_t)chunk * NROWS + row) >> 2;
    aggSum[q] = s;
    aggMax[q] = m;
    aggIdx[q] = am;
}

// =====================================================================
// Phase B: sequential exclusive prefix over the 64 chunk aggregates per
// row. Layout [chunk][row] -> each step a wave touches 32 consecutive
// rows = coalesced. 12288 threads total.
// =====================================================================
__global__ __launch_bounds__(256) void phaseB(float* __restrict__ aggSum,
                                              float* __restrict__ aggMax,
                                              int*   __restrict__ aggIdx) {
    const int row = blockIdx.x * blockDim.x + threadIdx.x;
    if (row >= NROWS) return;
    float s = 0.f, m = -INFINITY;
    int am = 0;
    for (int ch = 0; ch < NCH; ++ch) {
        const size_t idx = (size_t)ch * NROWS + row;
        const float cs = aggSum[idx];
        const float cm = aggMax[idx];
        const int   ci = aggIdx[idx];
        aggSum[idx] = s;          // overwrite with EXCLUSIVE prefix
        aggMax[idx] = m;
        aggIdx[idx] = am;
        s += cs;
        if (cm > m) { m = cm; am = ci; }   // strict > keeps earlier chunk on ties
    }
}

// =====================================================================
// Phase C: seeded rescan of each chunk + emit all outputs as B128
// non-temporal stores (~508 MB streamed, bypasses L2 residency).
// input-0 waves also broadcast the 20 MLP channels for their t-range.
// =====================================================================
__global__ __launch_bounds__(256) void phaseC(const float* __restrict__ i0,
                                              const float* __restrict__ i1,
                                              const float* __restrict__ i2,
                                              const f4* __restrict__ aggSum,
                                              const f4* __restrict__ aggMax,
                                              const i4* __restrict__ aggIdx,
                                              const float* __restrict__ d20,
                                              float* __restrict__ out,
                                              int*   __restrict__ inds,
                                              float* __restrict__ acts) {
    const int b   = blockIdx.x;
    const int cg  = b % (NCH / CGRP);
    const int n   = (b / (NCH / CGRP)) % NB;
    const int in  = b / ((NCH / CGRP) * NB);
    const int c4  = threadIdx.x & 31;
    const int chunk = cg * CGRP + (threadIdx.x >> 5);
    const int t0 = chunk * CH;

    const float* x = selInput(in, i0, i1, i2) + (size_t)n * LSEQ * CCH + c4 * 4;

    const int row = ((in * NB + n) * CCH + c4 * 4);
    const size_t q = ((size_t)chunk * NROWS + row) >> 2;
    f4 csum = aggSum[q];          // exclusive prefix seeds
    f4 cmax = aggMax[q];
    i4 amax = aggIdx[q];

    const size_t ioff  = ((size_t)in * NB + n) * LSEQ * CCH + c4 * 4;  // inds/acts [N,L,C]
    const size_t obase = (size_t)n * LSEQ * OUTC + in * 384 + c4 * 4;  // out [N,L,1172]
    const size_t dbase = (size_t)n * LSEQ * OUTC + 1152 + c4 * 4;
    const float invL = 1.0f / (float)LSEQ;

    const bool do_d = (in == 0) && (c4 < 5);   // 5 quads cover the 20 d-channels
    f4 dvals = {0.f, 0.f, 0.f, 0.f};
    if (do_d) dvals = *(const f4*)(d20 + n * 20 + c4 * 4);

    for (int tt = 0; tt < CH; ++tt) {
        const int t = t0 + tt;
        __builtin_prefetch(&x[(size_t)(t + 8) * CCH], 0, 1);
        const f4 v = *(const f4*)(x + (size_t)t * CCH);

        const bool  has_pad = (t < LSEQ - 1);      // window still contains pad zeros
        const float rc      = 1.0f / (float)(t + 1);
        f4 pmax, pavg, psum;
        i4 ind;
#pragma unroll
        for (int e = 0; e < 4; ++e) {
            if (v[e] > cmax[e]) { cmax[e] = v[e]; amax[e] = t; }
            csum[e] += v[e];
            pmax[e] = has_pad ? fmaxf(cmax[e], 0.f) : cmax[e];
            ind[e]  = (has_pad && cmax[e] < 0.f) ? (t - (LSEQ - 1)) : amax[e];
            pavg[e] = csum[e] * invL;
            psum[e] = csum[e] * rc;
        }

        const size_t o = obase + (size_t)t * OUTC;
        __builtin_nontemporal_store(pmax, (f4*)(out + o));            // in*384 + c
        __builtin_nontemporal_store(pavg, (f4*)(out + o + CCH));      // + 128
        __builtin_nontemporal_store(psum, (f4*)(out + o + 2 * CCH));  // + 256
        __builtin_nontemporal_store(ind,  (i4*)(inds + ioff + (size_t)t * CCH));
        __builtin_nontemporal_store(pmax, (f4*)(acts + ioff + (size_t)t * CCH));
        if (do_d)
            __builtin_nontemporal_store(dvals, (f4*)(out + dbase + (size_t)t * OUTC));
    }
}

// =====================================================================
extern "C" void kernel_launch(void* const* d_in, const int* in_sizes, int n_in,
                              void* d_out, int out_size, void* d_ws, size_t ws_size,
                              hipStream_t stream) {
    const float* inp0 = (const float*)d_in[0];
    const float* inp1 = (const float*)d_in[1];
    const float* inp2 = (const float*)d_in[2];
    const float* dem  = (const float*)d_in[3];
    const float* W1   = (const float*)d_in[4];
    const float* b1   = (const float*)d_in[5];
    const float* W2   = (const float*)d_in[6];
    const float* b2   = (const float*)d_in[7];

    float* out = (float*)d_out;
    // tuple layout: out [N,L,1172] | inds0..2 [N,L,C] i32 | acts0..2 [N,L,C] f32
    const size_t outElems = (size_t)NB * LSEQ * OUTC;
    const size_t planeSz  = (size_t)NB * LSEQ * CCH;
    int*   inds = (int*)(out + outElems);
    float* acts = out + outElems + 3 * planeSz;

    // workspace: d20 | aggSum | aggMax | aggIdx  ([chunk][row] layout)
    char* ws = (char*)d_ws;
    float* d20 = (float*)ws;
    const size_t AGG = (size_t)NROWS * NCH;       // 786432 entries each
    float* aggSum = (float*)(ws + 4096);
    float* aggMax = aggSum + AGG;
    int*   aggIdx = (int*)(aggMax + AGG);

    mlp_wmma<<<1, 32, 0, stream>>>(dem, W1, b1, W2, b2, d20);

    const int scanBlocks = NIN * NB * (NCH / CGRP);   // 768
    phaseA<<<scanBlocks, 256, 0, stream>>>(inp0, inp1, inp2,
                                           (f4*)aggSum, (f4*)aggMax, (i4*)aggIdx);
    phaseB<<<(NROWS + 255) / 256, 256, 0, stream>>>(aggSum, aggMax, aggIdx);
    phaseC<<<scanBlocks, 256, 0, stream>>>(inp0, inp1, inp2,
                                           (const f4*)aggSum, (const f4*)aggMax,
                                           (const i4*)aggIdx, d20, out, inds, acts);
}
